// BaseHSMM_17557826306367
// MI455X (gfx1250) — matrix operand order repaired and checked by reading, not compiled, not run on hardware
//
#include <hip/hip_runtime.h>
#include <math.h>

// ---------------------------------------------------------------------------
// Batched explicit-duration HSMM forward recursion for MI455X (gfx1250).
// One workgroup (256 threads = 8 wave32) per sequence; 32 WGs, one per WGP.
//
// Both O(K^2)-per-step reductions run on the matrix pipe in normalized
// exp-space (f16 operands in [0,1], f32 WMMA accumulation):
//   * transition:  trans_exp = exp(alpha-m) (1x128) @ exp(log_A) (128x128)
//   * duration:    dur[j] = diag( E_hist^T (16x128) @ [exp(log_D)*sc] (128x16) )
//     with per-row normalization R[s] baked into the f16 history ring and the
//     per-step global rebase sc[k] = exp(R[(t-k)&127] - M_old) folded into the
//     static W fragments (8 v_pk_mul_f16 per K-tile). sc[0]=0 masks the slot
//     being recycled (the expired d=128 row); the fresh d=0 term is added as
//     an exact scalar 2-way logsumexp. f16 underflow == pruning terms >16.6
//     log-units below the leader (numerically negligible ones).
// 4 split barriers per step; alpha / cumsum(log_B) register-resident.
// ---------------------------------------------------------------------------

typedef _Float16 v8h  __attribute__((ext_vector_type(8)));
typedef _Float16 v16h __attribute__((ext_vector_type(16)));
typedef float    v8f  __attribute__((ext_vector_type(8)));

#define HSMM_K     128
#define HSMM_DMAX  128
#define NEGBIG     (-1e30f)
#define NTHREADS   256

// LDS carve-up (bytes); total ~99KB of the 320KB WGP LDS.
#define OFF_EBUF   0          // _Float16 [128][128] exp(u - R[s]) ring   32KB
#define OFF_BF     32768      // v16h [8][4][32] exp(log_A)  B-frags      32KB
#define OFF_WF     65536      // v16h [8][4][32] exp(log_D)  B-frags      32KB
#define OFF_E16    98304      // _Float16 [128]  exp(alpha - m)
#define OFF_SC     98560      // _Float16 [128]  sc[k] rebase scales (32B al.)
#define OFF_R      98816      // float [128]     per-ring-row max R[s]
#define OFF_TEXP   99328      // float [128]     trans in exp space
#define OFF_DURX   99840      // float [128]     duration diag in exp space
#define OFF_RPART  100352     // float [4]  wave partials: masked R max
#define OFF_UPART  100416     // float [4]  wave partials: fresh-u max
#define OFF_WMAX   100480     // float [4]  wave partials: alpha max
#define OFF_AFIN   100544     // float [128] final alpha
#define SMEM_BYTES 101376

union AFrag { v16h v; v8h p[2]; _Float16 h[16]; };
union BFrag { v16h v; _Float16 h[16]; };

__device__ __forceinline__ float wave_max32(float v) {
#pragma unroll
  for (int off = 16; off; off >>= 1)
    v = fmaxf(v, __shfl_xor(v, off, 32));
  return v;
}

__global__ __launch_bounds__(NTHREADS)
void hsmm_forward(const float* __restrict__ log_B,
                  const float* __restrict__ log_pi,
                  const float* __restrict__ log_A,
                  const float* __restrict__ log_D,
                  float* __restrict__ out, int T) {
  extern __shared__ char smem[];
  _Float16* ebuf  = (_Float16*)(smem + OFF_EBUF);   // [s][j]
  v16h*     Bf    = (v16h*)(smem + OFF_BF);
  v16h*     Wf    = (v16h*)(smem + OFF_WF);
  _Float16* e16   = (_Float16*)(smem + OFF_E16);
  _Float16* sc16  = (_Float16*)(smem + OFF_SC);
  float*    R     = (float*)(smem + OFF_R);
  float*    texp  = (float*)(smem + OFF_TEXP);
  float*    durx  = (float*)(smem + OFF_DURX);
  float*    rpart = (float*)(smem + OFF_RPART);
  float*    upart = (float*)(smem + OFF_UPART);
  float*    wmaxs = (float*)(smem + OFF_WMAX);
  float*    afin  = (float*)(smem + OFF_AFIN);

  const int tid  = threadIdx.x;
  const int lane = tid & 31;
  const int wave = tid >> 5;                 // == j-tile for duration WMMA
  const int b    = blockIdx.x;
  const float* gB = log_B + (size_t)b * (size_t)T * HSMM_K;

  // -------- one-time setup --------
  // exp(log_A) and exp(log_D) packed into WMMA B-matrix lane layout:
  //   tile (nt,kt): lane L holds column N = nt*16 + (L&15),
  //   element i of v16h holds K = kt*32 + ((L&16)?16:0) + i.
  for (int e = tid; e < 8 * 4 * 32; e += NTHREADS) {
    int L = e & 31, kt = (e >> 5) & 3, nt = e >> 7;
    int kbase = kt * 32 + ((L & 16) ? 16 : 0);
    int j = nt * 16 + (L & 15);
    BFrag ba, bd;
#pragma unroll
    for (int i = 0; i < 16; ++i) {
      ba.h[i] = (_Float16)__expf(log_A[(kbase + i) * HSMM_K + j]);   // A[i->j]
      bd.h[i] = (_Float16)__expf(log_D[j * HSMM_DMAX + (kbase + i)]); // D[j][d]
    }
    Bf[e] = ba.v;
    Wf[e] = bd.v;
  }
  for (int i = tid; i < HSMM_DMAX * HSMM_K; i += NTHREADS) ebuf[i] = (_Float16)0.f;
  if (tid < HSMM_DMAX) R[tid] = NEGBIG;
  if (tid == 0) { wmaxs[0] = wmaxs[1] = wmaxs[2] = wmaxs[3] = NEGBIG; }

  float alpha = NEGBIG;                         // per-state register (tid < K)
  float cB    = 0.0f;                           // running cumsum(log_B[.,j])
  float gb    = (tid < HSMM_K) ? gB[tid] : 0.f; // pipelined log_B[t] row
  float d0reg = (tid < HSMM_K) ? log_D[tid * HSMM_DMAX] : 0.f; // log_D[j][0]
  __syncthreads();

  for (int t = 0; t < T; ++t) {
    const int tslot = t & 127;

    // ---- P1: alpha max -> normalized exp (f16); masked ring-row-max R ----
    float m = fmaxf(fmaxf(wmaxs[0], wmaxs[1]), fmaxf(wmaxs[2], wmaxs[3]));
    if (tid < HSMM_K) {
      e16[tid] = (_Float16)__expf(alpha - m);
      float rv = (tid == tslot) ? NEGBIG : R[tid];   // exclude expired slot
      rv = wave_max32(rv);
      if (lane == 0) rpart[tid >> 5] = rv;
    }
    __syncthreads();

    // ---- P2: global rebase M_old + sc[k]; transition WMMA -> texp ----
    float Mold = fmaxf(fmaxf(rpart[0], rpart[1]), fmaxf(rpart[2], rpart[3]));
    if (tid < HSMM_K) {
      float sv = (tid == 0) ? 0.0f : __expf(R[(t - tid) & 127] - Mold);
      sc16[tid] = (_Float16)sv;
    }
    {
      v8f c = {};
      const int kb0 = (lane & 16) ? 16 : 0;        // contiguous k = 16h + s, same as Bf packing
#pragma unroll
      for (int kt = 0; kt < 4; ++kt) {
        AFrag a;
        a.p[0] = *(const v8h*)&e16[kt * 32 + kb0];
        a.p[1] = *(const v8h*)&e16[kt * 32 + kb0 + 8];
        v16h bv = Bf[(wave * 4 + kt) * 32 + lane];
        c = __builtin_amdgcn_wmma_f32_16x16x32_f16(false, a.v, false, bv,
                                                   (short)0, c, false, false);
      }
      if (lane < 16) texp[wave * 16 + lane] = c[0]; // rows replicated
    }
    if (t + 8 < T && tid >= 128 && tid < 160)      // warm L2 (global_prefetch_b8)
      __builtin_prefetch(gB + (size_t)(t + 8) * HSMM_K + (tid - 128) * 4, 0, 0);
    __syncthreads();

    // ---- P3: scalar stream (start,u,cB) + duration WMMA diag ----
    float u = 0.0f;
    if (tid < HSMM_K) {
      float start = (t == 0) ? log_pi[tid]
                             : m + __logf(fmaxf(texp[tid], 1e-37f));
      u = start - cB;                              // u[t][j]
      cB += gb;                                    // now cB[t+1]
      if (t + 1 < T) gb = gB[(size_t)(t + 1) * HSMM_K + tid];
      float uw = wave_max32(u);
      if (lane == 0) upart[tid >> 5] = uw;
    }
    {
      v8f c = {};
      const int kb0 = (lane & 16) ? 16 : 0;
      const int jj  = (wave << 4) + (lane & 15);   // this lane's state row
#pragma unroll
      for (int kt = 0; kt < 4; ++kt) {
        AFrag a;                                   // A[m=j][k] = ebuf[(t-k)][j]
#pragma unroll
        for (int i = 0; i < 8; ++i) {
          int k0 = kt * 32 + kb0 + i;
          int k1 = k0 + 8;
          a.h[i]     = ebuf[((t - k0) & 127) * HSMM_K + jj];
          a.h[i + 8] = ebuf[((t - k1) & 127) * HSMM_K + jj];
        }
        v16h bw = Wf[(wave * 4 + kt) * 32 + lane];
        v16h sv = *(const v16h*)&sc16[kt * 32 + ((lane & 16) ? 16 : 0)];
        v16h bm = bw * sv;                         // v_pk_mul_f16 rebase fold
        c = __builtin_amdgcn_wmma_f32_16x16x32_f16(false, a.v, false, bm,
                                                   (short)0, c, false, false);
      }
      // diagonal extraction: lanes {0..7, 24..31} own diag element lane&7
      int rsel = lane & 7;
      float dv = c[0];
#pragma unroll
      for (int r = 1; r < 8; ++r) dv = (rsel == r) ? c[r] : dv;
      if ((lane < 8) || (lane >= 24)) durx[(wave << 4) + (lane & 15)] = dv;
    }
    __syncthreads();

    // ---- P4: write fresh ring row; alpha = cB + lse(dur_{d>=1}, d=0 term) --
    if (tid < HSMM_K) {
      float Rt = fmaxf(fmaxf(upart[0], upart[1]), fmaxf(upart[2], upart[3]));
      ebuf[tslot * HSMM_K + tid] = (_Float16)__expf(u - Rt);
      if (tid == 0) R[tslot] = Rt;
      float a1 = Mold + __logf(fmaxf(durx[tid], 1e-30f)); // durations 1..127
      float a2 = u + d0reg;                               // duration 0 (exact)
      float hi = fmaxf(a1, a2);
      alpha = cB + hi + __logf(__expf(a1 - hi) + __expf(a2 - hi));
      float wm = wave_max32(alpha);
      if (lane == 0) wmaxs[tid >> 5] = wm;
    }
    __syncthreads();
  }

  // -------- final logsumexp over states --------
  if (tid < HSMM_K) afin[tid] = alpha;
  __syncthreads();
  if (tid == 0) {
    float mf = NEGBIG;
    for (int j = 0; j < HSMM_K; ++j) mf = fmaxf(mf, afin[j]);
    float sf = 0.0f;
    for (int j = 0; j < HSMM_K; ++j) sf += __expf(afin[j] - mf);
    out[b] = mf + __logf(sf);
  }
}

extern "C" void kernel_launch(void* const* d_in, const int* in_sizes, int n_in,
                              void* d_out, int out_size, void* d_ws, size_t ws_size,
                              hipStream_t stream) {
  (void)d_ws; (void)ws_size; (void)n_in;
  const float* log_B  = (const float*)d_in[0];
  const float* log_pi = (const float*)d_in[1];
  const float* log_A  = (const float*)d_in[2];
  const float* log_D  = (const float*)d_in[3];
  float* out = (float*)d_out;

  const int B = out_size;
  const int T = in_sizes[0] / (HSMM_K * (B > 0 ? B : 1));

  hipFuncSetAttribute((const void*)hsmm_forward,
                      hipFuncAttributeMaxDynamicSharedMemorySize, SMEM_BYTES);

  hipLaunchKernelGGL(hsmm_forward, dim3(B), dim3(NTHREADS), SMEM_BYTES, stream,
                     log_B, log_pi, log_A, log_D, out, T);
}
